// MultiHeadAttentionGPT_49709951484072
// MI455X (gfx1250) — compile-verified, matrix operand order checked by reading, not run on hardware
//
#include <hip/hip_runtime.h>
#include <hip/hip_bf16.h>

typedef __bf16 bf16;
typedef __attribute__((ext_vector_type(16))) __bf16 v16bf;
typedef __attribute__((ext_vector_type(8)))  __bf16 v8bf;
typedef __attribute__((ext_vector_type(8)))  float  v8f;

union F8 { v8f v; float f[8]; };

#define BATCH 4
#define SEQ 2048
#define DMODEL 1024
#define NHEADS 16
#define DHEAD 64

static __device__ __forceinline__ v16bf load2x8(const bf16* p0, const bf16* p1) {
    union { v16bf v; v8bf h[2]; } u;
    u.h[0] = *reinterpret_cast<const v8bf*>(p0);
    u.h[1] = *reinterpret_cast<const v8bf*>(p1);
    return u.v;
}

// 16x32 bf16 fragment (A-layout) from a row-major [*, ld] bf16 matrix at (row0, k0).
// Lane l holds row (l&15); K elems: [k0 + 8*(l>>4), +8) and [k0 + 16 + 8*(l>>4), +8).
// The B-fragment of a KxN matrix uses the same pattern on its transpose ([N,K] row-major).
static __device__ __forceinline__ v16bf frag_a(const bf16* base, int row0, int k0, int ld, int lane) {
    int r  = lane & 15;
    int hk = lane >> 4;
    const bf16* p = base + (size_t)(row0 + r) * ld + k0 + 8 * hk;
    return load2x8(p, p + 16);
}

static __device__ __forceinline__ v8f wmma_bf16(v16bf a, v16bf b, v8f c) {
    return __builtin_amdgcn_wmma_f32_16x16x32_bf16(
        /*neg_a=*/false, a, /*neg_b=*/false, b,
        /*c_mod=*/(short)0, c, /*reuse_a=*/false, /*reuse_b=*/false);
}

// ---------------- conversion kernels ----------------

__global__ void cvt_f32_to_bf16_kernel(const float* __restrict__ src,
                                       bf16* __restrict__ dst, size_t n) {
    size_t i = (size_t)blockIdx.x * blockDim.x + threadIdx.x;
    size_t stride = (size_t)gridDim.x * blockDim.x;
    for (; i < n; i += stride) dst[i] = (bf16)src[i];
}

// src [rows][cols] fp32 -> dst [cols][rows] bf16
__global__ void transpose_f32_to_bf16_kernel(const float* __restrict__ src,
                                             bf16* __restrict__ dst,
                                             int rows, int cols) {
    int i = blockIdx.x * blockDim.x + threadIdx.x;
    if (i < rows * cols) {
        int r = i / cols, c = i % cols;
        dst[(size_t)c * rows + r] = (bf16)src[i];
    }
}

// ---------------- bf16 GEMM: C = A[M,K] * B (B given transposed: BT[N,K]) ----------------
// Wave tile 64x32 (4 M-tiles x 2 N-tiles) -> 8 WMMAs per 6 fragment loads per K=32 step.
// 8 waves arranged 2(M) x 4(N): macro tile 128x128.
// MODE 0: out fp32 [M,N] row-major
// MODE 1: out bf16 head-split   [B, H, seq, DHEAD]   (Q, K)
// MODE 2: out bf16 head-split-T [B, H, DHEAD, seq]   (V transposed for P*V)

template <int MODE>
__global__ __launch_bounds__(256) void gemm_bf16_kernel(
    const bf16* __restrict__ A, const bf16* __restrict__ BT,
    void* __restrict__ out, int M, int N, int K)
{
    int lane = threadIdx.x & 31;
    int wave = threadIdx.x >> 5;      // 8 waves
    int wm = wave & 1;                // 2 waves along M
    int wn = wave >> 1;               // 4 waves along N
    int m0 = blockIdx.x * 128 + wm * 64;
    int n0 = blockIdx.y * 128 + wn * 32;

    v8f c[4][2] = {};
    for (int k = 0; k < K; k += 32) {
        v16bf b0 = frag_a(BT, n0,      k, K, lane);
        v16bf b1 = frag_a(BT, n0 + 16, k, K, lane);
        #pragma unroll
        for (int tm = 0; tm < 4; ++tm) {
            v16bf a0 = frag_a(A, m0 + tm * 16, k, K, lane);
            c[tm][0] = wmma_bf16(a0, b0, c[tm][0]);
            c[tm][1] = wmma_bf16(a0, b1, c[tm][1]);
        }
    }

    int half = lane >> 4, nl = lane & 15;
    #pragma unroll
    for (int tm = 0; tm < 4; ++tm)
    #pragma unroll
    for (int tn = 0; tn < 2; ++tn) {
        F8 cc; cc.v = c[tm][tn];
        #pragma unroll
        for (int v = 0; v < 8; ++v) {
            int gm = m0 + tm * 16 + v + 8 * half;   // row in [M)
            int gn = n0 + tn * 16 + nl;             // col in [N)
            float val = cc.f[v];
            if (MODE == 0) {
                ((float*)out)[(size_t)gm * N + gn] = val;
            } else {
                int b = gm >> 11, seq = gm & (SEQ - 1);
                int h = gn >> 6,  d   = gn & (DHEAD - 1);
                bf16* dst = (bf16*)out;
                if (MODE == 1)
                    dst[(((size_t)b * NHEADS + h) * SEQ + seq) * DHEAD + d] = (bf16)val;
                else
                    dst[(((size_t)b * NHEADS + h) * DHEAD + d) * SEQ + seq] = (bf16)val;
            }
        }
    }
}

// ---------------- causal flash attention ----------------
// Q: [B,H,SEQ,DHEAD] bf16,  Kc: [B,H,SEQ,DHEAD] bf16 (rows = keys, serves as K^T B-frag),
// Vt: [B,H,DHEAD,SEQ] bf16, ctx out: [B*SEQ, DMODEL] bf16
// One wave per 16-query block; key blocks of 32 (one WMMA K-step for P*V).

__global__ __launch_bounds__(128) void attn_causal_kernel(
    const bf16* __restrict__ Q, const bf16* __restrict__ Kc,
    const bf16* __restrict__ Vt, bf16* __restrict__ ctx)
{
    __shared__ __align__(16) bf16 pshare[4][16 * 32];

    int lane = threadIdx.x & 31;
    int wave = threadIdx.x >> 5;
    int qblk = blockIdx.x * 4 + wave;   // 0..127
    int bh   = blockIdx.y;              // 0..63
    int b = bh >> 4, h = bh & 15;

    const bf16* q  = Q  + (size_t)bh * SEQ * DHEAD;
    const bf16* kk = Kc + (size_t)bh * SEQ * DHEAD;
    const bf16* vt = Vt + (size_t)bh * DHEAD * SEQ;

    int q0 = qblk * 16;
    int half = lane >> 4, nl = lane & 15;

    // Q fragments: 16 rows x 64 d  ->  two K=32 chunks
    v16bf qf0 = frag_a(q, q0, 0,  DHEAD, lane);
    v16bf qf1 = frag_a(q, q0, 32, DHEAD, lane);

    F8 o[4] = {};                       // 16 x 64 fp32 accumulator (4 N-tiles)
    float mrow[8], lrow[8];
    #pragma unroll
    for (int i = 0; i < 8; ++i) { mrow[i] = -__builtin_inff(); lrow[i] = 0.f; }

    const float scale = 0.125f;         // 1/sqrt(64)
    int kb_last = (q0 + 15) >> 5;

    for (int kb = 0; kb <= kb_last; ++kb) {
        int k0 = kb * 32;

        // S = Q * K^T : two 16-key tiles, K-dim 64 split in 2x32
        F8 s0, s1;
        s0.v = (v8f){}; s1.v = (v8f){};
        s0.v = wmma_bf16(qf0, frag_a(kk, k0,      0,  DHEAD, lane), s0.v);
        s0.v = wmma_bf16(qf1, frag_a(kk, k0,      32, DHEAD, lane), s0.v);
        s1.v = wmma_bf16(qf0, frag_a(kk, k0 + 16, 0,  DHEAD, lane), s1.v);
        s1.v = wmma_bf16(qf1, frag_a(kk, k0 + 16, 32, DHEAD, lane), s1.v);

        bool diag = (kb == kb_last);
        bf16* ps = pshare[wave];

        #pragma unroll
        for (int v = 0; v < 8; ++v) {
            int row = q0 + v + 8 * half;
            float a  = s0.f[v] * scale;
            float bb = s1.f[v] * scale;
            if (diag) {
                if (k0 + nl      > row) a  = -__builtin_inff();
                if (k0 + 16 + nl > row) bb = -__builtin_inff();
            }
            // row max over the 16 lanes of this half-wave
            float mx = fmaxf(a, bb);
            #pragma unroll
            for (int off = 8; off >= 1; off >>= 1)
                mx = fmaxf(mx, __shfl_xor(mx, off, 32));
            float mnew  = fmaxf(mrow[v], mx);
            float alpha = __expf(mrow[v] - mnew);
            float ea = __expf(a  - mnew);
            float eb = __expf(bb - mnew);
            float rs = ea + eb;
            #pragma unroll
            for (int off = 8; off >= 1; off >>= 1)
                rs += __shfl_xor(rs, off, 32);
            lrow[v] = lrow[v] * alpha + rs;
            mrow[v] = mnew;
            #pragma unroll
            for (int t = 0; t < 4; ++t) o[t].f[v] *= alpha;
            // stage P (bf16) in LDS [16 rows][32 keys] (C-layout -> A-layout shuffle)
            int r = v + 8 * half;
            ps[r * 32 + nl]      = (bf16)ea;
            ps[r * 32 + 16 + nl] = (bf16)eb;
        }
        // same-wave DS ops are in-order (ISA 7.3) -> no barrier needed
        v16bf pf = frag_a(ps, 0, 0, 32, lane);

        // O += P * V  (V^T layout: contiguous in key dim)
        #pragma unroll
        for (int t = 0; t < 4; ++t) {
            v16bf vf = frag_a(vt, t * 16, k0, SEQ, lane);
            o[t].v = wmma_bf16(pf, vf, o[t].v);
        }
    }

    // finalize: divide by row sums, write ctx [b, seq, h*64 + d] as bf16
    #pragma unroll
    for (int v = 0; v < 8; ++v) {
        float inv = 1.f / lrow[v];
        int seq = q0 + v + 8 * half;
        #pragma unroll
        for (int t = 0; t < 4; ++t) {
            float val = o[t].f[v] * inv;
            ctx[((size_t)b * SEQ + seq) * DMODEL + h * DHEAD + t * 16 + nl] = (bf16)val;
        }
    }
}

// ---------------- launcher ----------------

extern "C" void kernel_launch(void* const* d_in, const int* in_sizes, int n_in,
                              void* d_out, int out_size, void* d_ws, size_t ws_size,
                              hipStream_t stream) {
    const float* X  = (const float*)d_in[0];
    const float* Wq = (const float*)d_in[1];
    const float* Wk = (const float*)d_in[2];
    const float* Wv = (const float*)d_in[3];
    const float* Wo = (const float*)d_in[4];
    float* out = (float*)d_out;

    const size_t M = (size_t)BATCH * SEQ;                 // 8192
    const size_t xElems = M * DMODEL;                     // 8.39M
    const size_t wElems = (size_t)DMODEL * DMODEL;        // 1.05M

    char* ws = (char*)d_ws;
    bf16* xb   = (bf16*)ws; ws += xElems * sizeof(bf16);
    bf16* wqT  = (bf16*)ws; ws += wElems * sizeof(bf16);
    bf16* wkT  = (bf16*)ws; ws += wElems * sizeof(bf16);
    bf16* wvT  = (bf16*)ws; ws += wElems * sizeof(bf16);
    bf16* woT  = (bf16*)ws; ws += wElems * sizeof(bf16);
    bf16* qh   = (bf16*)ws; ws += xElems * sizeof(bf16);  // [B,H,S,D]
    bf16* kh   = (bf16*)ws; ws += xElems * sizeof(bf16);  // [B,H,S,D]
    bf16* vth  = (bf16*)ws; ws += xElems * sizeof(bf16);  // [B,H,D,S]
    bf16* ctxb = (bf16*)ws; ws += xElems * sizeof(bf16);  // [M, DMODEL]

    // 1) precision conversion / weight transposes
    cvt_f32_to_bf16_kernel<<<4096, 256, 0, stream>>>(X, xb, xElems);
    transpose_f32_to_bf16_kernel<<<(int)(wElems / 256), 256, 0, stream>>>(Wq, wqT, DMODEL, DMODEL);
    transpose_f32_to_bf16_kernel<<<(int)(wElems / 256), 256, 0, stream>>>(Wk, wkT, DMODEL, DMODEL);
    transpose_f32_to_bf16_kernel<<<(int)(wElems / 256), 256, 0, stream>>>(Wv, wvT, DMODEL, DMODEL);
    transpose_f32_to_bf16_kernel<<<(int)(wElems / 256), 256, 0, stream>>>(Wo, woT, DMODEL, DMODEL);

    // 2) QKV projections (WMMA GEMM, head-split epilogues); macro tile 128x128
    dim3 ggrid((unsigned)(M / 128), DMODEL / 128);
    gemm_bf16_kernel<1><<<ggrid, 256, 0, stream>>>(xb, wqT, qh,  (int)M, DMODEL, DMODEL);
    gemm_bf16_kernel<1><<<ggrid, 256, 0, stream>>>(xb, wkT, kh,  (int)M, DMODEL, DMODEL);
    gemm_bf16_kernel<2><<<ggrid, 256, 0, stream>>>(xb, wvT, vth, (int)M, DMODEL, DMODEL);

    // 3) causal flash attention: 128 qblocks / 4 waves per WG, 64 (batch,head) pairs
    dim3 agrid(SEQ / 16 / 4, BATCH * NHEADS);
    attn_causal_kernel<<<agrid, 128, 0, stream>>>(qh, kh, vth, ctxb);

    // 4) output projection -> fp32
    gemm_bf16_kernel<0><<<ggrid, 256, 0, stream>>>(ctxb, woT, out, (int)M, DMODEL, DMODEL);
}
